// SinkhornAttn_56530359550027
// MI455X (gfx1250) — matrix operand order repaired and checked by reading, not compiled, not run on hardware
//
#include <hip/hip_runtime.h>
#include <hip/hip_bf16.h>
#include <cstdint>

typedef __attribute__((ext_vector_type(16))) _Float16 v16h;
typedef __attribute__((ext_vector_type(8)))  float    v8f;

#define NUM_BH    64            // b(4) * H(16)
#define BUCKETS   32
#define BSZ       128           // bucket size
#define DIM       64            // head dim
#define TILE_ELEMS (BSZ * DIM)  // 8192 floats per (bh,u) tile
#define LDQK      72            // padded f16 leading dim for Q/K/V tiles
#define LDP       136           // padded f16 leading dim for P (128x128)

// ---------------------------------------------------------------------------
// Kernel 1: pl[bh,u,d] = sum_s K[bh,u,s,d]   (bucket summaries)
// ---------------------------------------------------------------------------
__global__ __launch_bounds__(64)
void SA_plsum_kernel(const float* __restrict__ k, float* __restrict__ pl) {
    const int tile = blockIdx.x;          // bh*32 + u, 2048 blocks
    const int d    = threadIdx.x;         // 0..63
    const float* base = k + (size_t)tile * TILE_ELEMS + d;
    float acc = 0.f;
#pragma unroll 8
    for (int s = 0; s < BSZ; ++s) acc += base[s * DIM];
    pl[tile * DIM + d] = acc;
}

// ---------------------------------------------------------------------------
// Kernel 2: routing. One wave per (b,h): logits = pl @ W, + gumbel, 5 sinkhorn
// iterations (row/col logsumexp), exp -> R[bh,32,32].
// ---------------------------------------------------------------------------
__device__ __forceinline__ float SA_uniform(uint32_t x) {
    // deterministic hash RNG standing in for jax.random (key=42)
    x = x * 0x9E3779B9u + 42u * 0x7F4A7C15u;
    x ^= x >> 16; x *= 0x85EBCA6Bu;
    x ^= x >> 13; x *= 0xC2B2AE35u;
    x ^= x >> 16;
    return (float)(x >> 8) * (1.0f / 16777216.0f);
}

__global__ __launch_bounds__(32)
void SA_route_kernel(const float* __restrict__ pl, const float* __restrict__ Wk,
                     float* __restrict__ Rout) {
    const int bh = blockIdx.x;            // 0..63
    const int h  = bh & 15;
    const int i  = threadIdx.x;           // row of R, 0..31
    __shared__ float lds[32 * 33];
    __shared__ float col_lse[32];

    float pld[DIM];
#pragma unroll
    for (int d = 0; d < DIM; ++d) pld[d] = pl[(bh * 32 + i) * DIM + d];

    float r[32];
#pragma unroll
    for (int f = 0; f < 32; ++f) {
        float s = 0.f;
#pragma unroll
        for (int d = 0; d < DIM; ++d) s += pld[d] * Wk[(h * DIM + d) * 32 + f];
        const float u = SA_uniform((uint32_t)((bh * 32 + i) * 32 + f));
        const float g = -__logf(-__logf(u + 1e-6f) + 1e-6f);
        r[f] = (s + g) * (1.0f / 0.75f);    // TEMP = 0.75
    }

    for (int it = 0; it < 5; ++it) {
        // row logsumexp (lane-local)
        float m = -3.0e38f;
#pragma unroll
        for (int f = 0; f < 32; ++f) m = fmaxf(m, r[f]);
        float sm = 0.f;
#pragma unroll
        for (int f = 0; f < 32; ++f) sm += __expf(r[f] - m);
        const float lse = m + __logf(sm);
#pragma unroll
        for (int f = 0; f < 32; ++f) { r[f] -= lse; lds[i * 33 + f] = r[f]; }
        __syncthreads();
        // column logsumexp: lane i reduces column i
        float cm = -3.0e38f;
#pragma unroll
        for (int ii = 0; ii < 32; ++ii) cm = fmaxf(cm, lds[ii * 33 + i]);
        float cs = 0.f;
#pragma unroll
        for (int ii = 0; ii < 32; ++ii) cs += __expf(lds[ii * 33 + i] - cm);
        col_lse[i] = cm + __logf(cs);
        __syncthreads();
#pragma unroll
        for (int f = 0; f < 32; ++f) r[f] -= col_lse[f];
        __syncthreads();
    }
#pragma unroll
    for (int f = 0; f < 32; ++f) Rout[(bh * 32 + i) * 32 + f] = __expf(r[f]);
}

// ---------------------------------------------------------------------------
// Kernel 3: Kmix[bh,i,:] = sum_j (R[i,j] + delta_ij) * K[bh,j,:]
// Staged directly into d_out (attention block (bh,u) reads exactly the tile it
// later overwrites, so no cross-block hazard). Memory-bound: 128 MB total.
// ---------------------------------------------------------------------------
__global__ __launch_bounds__(256)
void SA_kmix_kernel(const float* __restrict__ k, const float* __restrict__ R,
                    float* __restrict__ kmix) {
    const int bh = blockIdx.y;
    const int cc = blockIdx.x * 256 + threadIdx.x;   // 0..8191 = s*64+d
    __shared__ float Rs[BUCKETS * BUCKETS];
    for (int idx = threadIdx.x; idx < BUCKETS * BUCKETS; idx += 256) {
        const int ii = idx >> 5, jj = idx & 31;
        Rs[idx] = R[bh * 1024 + idx] + ((ii == jj) ? 1.0f : 0.0f);
    }
    __syncthreads();

    const float* kb = k + (size_t)bh * BUCKETS * TILE_ELEMS + cc;
    float acc[BUCKETS];
#pragma unroll
    for (int i2 = 0; i2 < BUCKETS; ++i2) acc[i2] = 0.f;
#pragma unroll 4
    for (int j = 0; j < BUCKETS; ++j) {
        const float kj = kb[(size_t)j * TILE_ELEMS];
#pragma unroll
        for (int i2 = 0; i2 < BUCKETS; ++i2) acc[i2] += Rs[i2 * 32 + j] * kj;
    }
    float* ob = kmix + (size_t)bh * BUCKETS * TILE_ELEMS + cc;
#pragma unroll
    for (int i2 = 0; i2 < BUCKETS; ++i2) ob[(size_t)i2 * TILE_ELEMS] = acc[i2];
}

// ---------------------------------------------------------------------------
// Kernel 4: per-bucket attention with WMMA.
//   S = Q Kmix^T / 8 ; P = softmax(S) ; O = P V
// 8 wave32s per block; wave w owns rows 16w..16w+15.
// LDS: Q[128x72] | K[128x72] | V[128x72] f16; P[128x136] aliases Q+K region.
// ---------------------------------------------------------------------------
__global__ __launch_bounds__(256)
void SA_attn_kernel(const float* __restrict__ q, const float* __restrict__ kmix,
                    const float* __restrict__ v, float* __restrict__ out) {
    const int tile = blockIdx.x;                  // bh*32 + u, 2048 blocks
    const size_t base = (size_t)tile * TILE_ELEMS;
    const int tid  = threadIdx.x;
    const int lane = tid & 31;
    const int wave = tid >> 5;                    // M tile index 0..7
    const int nIdx = lane & 15;
    const int kg   = lane >> 4;                   // which 16-lane half

    __shared__ __align__(16) _Float16 smem[3 * BSZ * LDQK];  // 55296 B
    _Float16* lds_q = smem;
    _Float16* lds_k = smem + BSZ * LDQK;
    _Float16* lds_v = smem + 2 * BSZ * LDQK;
    _Float16* lds_p = smem;                       // aliases Q+K after S phase

    // ---- stage tiles (fp32 -> f16), vectorized/coalesced ----
    {
        const float4* q4 = (const float4*)(q + base);
        const float4* k4 = (const float4*)(kmix + base);
        const float4* v4 = (const float4*)(v + base);
        for (int i = tid; i < TILE_ELEMS / 4; i += 256) {
            const int row = i >> 4;
            const int col = (i & 15) << 2;
            const float4 a = q4[i];
            const float4 b = k4[i];
            const float4 c = v4[i];
            _Float16* dq = &lds_q[row * LDQK + col];
            dq[0] = (_Float16)a.x; dq[1] = (_Float16)a.y;
            dq[2] = (_Float16)a.z; dq[3] = (_Float16)a.w;
            _Float16* dk = &lds_k[row * LDQK + col];
            dk[0] = (_Float16)b.x; dk[1] = (_Float16)b.y;
            dk[2] = (_Float16)b.z; dk[3] = (_Float16)b.w;
            _Float16* dv = &lds_v[row * LDQK + col];
            dv[0] = (_Float16)c.x; dv[1] = (_Float16)c.y;
            dv[2] = (_Float16)c.z; dv[3] = (_Float16)c.w;
        }
    }
    __syncthreads();

    // ---- Q A-fragments (16x32 f16 layout: lane m = lane&15; K per element) --
    v16h aQ0, aQ1;
    {
        const _Float16* qrow = &lds_q[(16 * wave + nIdx) * LDQK];
#pragma unroll
        for (int e = 0; e < 16; ++e) {
            const int ke = (e & 7) + 8 * kg + 16 * (e >> 3);
            aQ0[e] = qrow[ke];
            aQ1[e] = qrow[32 + ke];
        }
    }

    // ---- S = Q Kmix^T : 8 N-tiles, K=64 -> 2 wmma each ----
    v8f acc[8];
#pragma unroll
    for (int t = 0; t < 8; ++t) {
        v16h b0, b1;
        const _Float16* krow = &lds_k[(16 * t + nIdx) * LDQK];
#pragma unroll
        for (int e = 0; e < 16; ++e) {
            const int ke = (e & 7) + 8 * kg + 16 * (e >> 3);          // B 32x16 layout: K = (e&7) + 8*kg + 16*(e>>3)
            b0[e] = krow[ke];
            b1[e] = krow[32 + ke];
        }
        v8f c = {};
        c = __builtin_amdgcn_wmma_f32_16x16x32_f16(false, aQ0, false, b0,
                                                   (short)0, c, false, false);
        c = __builtin_amdgcn_wmma_f32_16x16x32_f16(false, aQ1, false, b1,
                                                   (short)0, c, false, false);
        acc[t] = c;
    }

    // ---- row softmax in registers (rows split across 16-lane halves) ----
    float inv[8];
#pragma unroll
    for (int r = 0; r < 8; ++r) {
        float m = -3.0e38f;
#pragma unroll
        for (int t = 0; t < 8; ++t) {
            acc[t][r] *= 0.125f;                 // d^-0.5
            m = fmaxf(m, acc[t][r]);
        }
#pragma unroll
        for (int off = 1; off < 16; off <<= 1) m = fmaxf(m, __shfl_xor(m, off, 32));
        float s = 0.f;
#pragma unroll
        for (int t = 0; t < 8; ++t) {
            const float p = __expf(acc[t][r] - m);
            acc[t][r] = p;
            s += p;
        }
#pragma unroll
        for (int off = 1; off < 16; off <<= 1) s += __shfl_xor(s, off, 32);
        inv[r] = 1.0f / s;
    }

    __syncthreads();   // all waves finished reading lds_q / lds_k
#pragma unroll
    for (int r = 0; r < 8; ++r) {
        const int row = 16 * wave + r + 8 * kg;
#pragma unroll
        for (int t = 0; t < 8; ++t)
            lds_p[row * LDP + 16 * t + nIdx] = (_Float16)(acc[t][r] * inv[r]);
    }
    __syncthreads();

    // ---- O = P V : K=128 -> 4 wmma per 16-wide N tile (4 tiles of d) ----
    v16h aP[4];
    {
        const _Float16* prow = &lds_p[(16 * wave + nIdx) * LDP];
#pragma unroll
        for (int kk = 0; kk < 4; ++kk)
#pragma unroll
            for (int e = 0; e < 16; ++e) {
                const int ke = 32 * kk + (e & 7) + 8 * kg + 16 * (e >> 3);
                aP[kk][e] = prow[ke];
            }
    }
#pragma unroll
    for (int t = 0; t < 4; ++t) {
        v8f c = {};
#pragma unroll
        for (int kk = 0; kk < 4; ++kk) {
            v16h bV;
#pragma unroll
            for (int e = 0; e < 16; ++e) {
                const int j = 32 * kk + (e & 7) + 8 * kg + 16 * (e >> 3);   // token index (K dim)
                bV[e] = lds_v[j * LDQK + 16 * t + nIdx];
            }
            c = __builtin_amdgcn_wmma_f32_16x16x32_f16(false, aP[kk], false, bV,
                                                       (short)0, c, false, false);
        }
        float* ob = out + base;
#pragma unroll
        for (int r = 0; r < 8; ++r)
            ob[(16 * wave + r + 8 * kg) * DIM + 16 * t + nIdx] = c[r];
    }
}

// ---------------------------------------------------------------------------
extern "C" void kernel_launch(void* const* d_in, const int* in_sizes, int n_in,
                              void* d_out, int out_size, void* d_ws, size_t ws_size,
                              hipStream_t stream) {
    (void)in_sizes; (void)n_in; (void)out_size; (void)ws_size;
    const float* q  = (const float*)d_in[0];
    const float* k  = (const float*)d_in[1];
    const float* v  = (const float*)d_in[2];
    const float* Wk = (const float*)d_in[3];
    // d_in[4] (W_vals) is dead in the reference output path.
    float* out = (float*)d_out;

    float* pl = (float*)d_ws;            // 2048 * 64 floats  (512 KB)
    float* R  = pl + 2048 * 64;          // 64 * 32 * 32 floats (256 KB)

    SA_plsum_kernel<<<2048, 64, 0, stream>>>(k, pl);
    SA_route_kernel<<<64, 32, 0, stream>>>(pl, Wk, R);
    // Stage Kmix in d_out; attention blocks read exactly the tile they overwrite.
    SA_kmix_kernel<<<dim3(32, NUM_BH), 256, 0, stream>>>(k, R, out);
    SA_attn_kernel<<<2048, 256, 0, stream>>>(q, out, v, out);
}
